// MultiHeadDifferentialAttention_68753836474706
// MI455X (gfx1250) — matrix operand order repaired and checked by reading, not compiled, not run on hardware
//
#include <hip/hip_runtime.h>
#include <hip/hip_bf16.h>
#include <cstdint>
#include <cstddef>

// Problem constants (from reference)
#define B_  2
#define S_  2048
#define D_  1024
#define H_  16
#define DH_ 64
#define SCALE_ 0.125f            // 1/sqrt(64)
#define LAMBDA_INIT_ 0.8f

typedef __attribute__((ext_vector_type(16))) _Float16 v16h;
typedef __attribute__((ext_vector_type(8)))  _Float16 v8h;
typedef __attribute__((ext_vector_type(8)))  float    v8f;

__device__ __forceinline__ v16h pack16(v8h lo, v8h hi) {
  return __builtin_shufflevector(lo, hi, 0,1,2,3,4,5,6,7,8,9,10,11,12,13,14,15);
}

__device__ __forceinline__ v8f wmma_f16(v16h a, v16h b, v8f c) {
  // D = A(16x32 f16) * B(32x16 f16) + C(16x16 f32)
  return __builtin_amdgcn_wmma_f32_16x16x32_f16(false, a, false, b, (short)0, c, false, false);
}

// A-fragment (16x32) from a row-major f16 tile with leading dim `lda` (halves).
// Lane layout: lanes 0-15 -> M=lane, K 0..15; lanes 16-31 -> K 16..31.
__device__ __forceinline__ v16h load_a_frag(const _Float16* base, int lda) {
  const int lane = threadIdx.x & 31;
  const int row = lane & 15, hi = lane >> 4;
  const _Float16* p = base + row * lda + 16 * hi;
  v8h lo = *(const v8h*)(p);
  v8h hv = *(const v8h*)(p + 8);
  return pack16(lo, hv);
}

// B-fragment (32x16) from a row-major [k][n] f16 tile, leading dim `ldb`.
// lane = column; lanes 16-31 cover K=16..31.
__device__ __forceinline__ v16h load_b_frag(const _Float16* base, int ldb) {
  const int lane = threadIdx.x & 31;
  const int col = lane & 15, hi = lane >> 4;
  const _Float16* p = base + (hi * 16) * ldb + col;
  v16h r;
#pragma unroll
  for (int e = 0; e < 16; ++e) r[e] = p[e * ldb];
  return r;
}

// B-fragment where the source matrix is stored transposed, i.e. the 16
// K-elements a lane needs are contiguous at `p` (caller folds col & hi in).
__device__ __forceinline__ v16h load_b_frag_contig(const _Float16* p) {
  v8h lo = *(const v8h*)(p);
  v8h hv = *(const v8h*)(p + 8);
  return pack16(lo, hv);
}

// --------------------------------------------------------------------------
// lambda[h] = exp(lq1.lk1) - exp(lq2.lk2) + 0.8
__global__ void lam_kernel(const float* __restrict__ lq1, const float* __restrict__ lk1,
                           const float* __restrict__ lq2, const float* __restrict__ lk2,
                           float* __restrict__ lam) {
  int h = threadIdx.x;
  if (h < H_) {
    float d1 = 0.f, d2 = 0.f;
    for (int i = 0; i < DH_; ++i) {
      d1 += lq1[h * DH_ + i] * lk1[h * DH_ + i];
      d2 += lq2[h * DH_ + i] * lk2[h * DH_ + i];
    }
    lam[h] = __expf(d1) - __expf(d2) + LAMBDA_INIT_;
  }
}

// --------------------------------------------------------------------------
// Tiled WMMA GEMM: C = A(MxK) @ W(KxN).  BM=128, BN=128, BK=32, 8 waves in a
// 4(M) x 2(N) grid; each wave computes 32x64 = 8 WMMA tiles per K-step, so the
// compute-to-LDS-staging ratio is 4x the naive 64x64 version.
// MODE 0: A f32, store f16 to [b][h][s][128]   (QK projections, N=2048)
// MODE 1: A f32, store f16 to [b][h][d][s]     (V projection transposed, N=1024)
// MODE 2: A f16, store f32 row-major           (output projection)
template <int MODE>
__launch_bounds__(256, 2)
__global__ void gemm_kernel(const void* __restrict__ Av, const float* __restrict__ W,
                            void* __restrict__ Cv, int M, int N, int K) {
  __shared__ __align__(16) _Float16 Asm[128][40];   // 128x32 + pad
  __shared__ __align__(16) _Float16 Bsm[32][136];   // 32x128 + pad

  const int tid  = threadIdx.x;
  const int lane = tid & 31;
  const int wave = tid >> 5;
  const int wm = wave & 3;             // 32-row subtile group
  const int wn = wave >> 2;            // 64-col subtile group
  const int mBase = blockIdx.y * 128;
  const int nBase = blockIdx.x * 128;

  v8f acc[2][4];
#pragma unroll
  for (int mi = 0; mi < 2; ++mi)
#pragma unroll
    for (int ni = 0; ni < 4; ++ni) acc[mi][ni] = (v8f){};

  const int ar = tid >> 1, ac = (tid & 1) * 16;   // A staging: 16 elems each
  const int wr = tid >> 3, wc = (tid & 7) * 16;   // W staging: 16 elems each

  for (int k0 = 0; k0 < K; k0 += 32) {
    __syncthreads();
    // ---- stage A tile (128x32) as f16
    if (MODE == 2) {
      const _Float16* A = (const _Float16*)Av;
      const _Float16* ap = A + (size_t)(mBase + ar) * K + k0 + ac;
      *(v8h*)(&Asm[ar][ac])     = *(const v8h*)(ap);
      *(v8h*)(&Asm[ar][ac + 8]) = *(const v8h*)(ap + 8);
    } else {
      const float* A = (const float*)Av;
      const float* ap = A + (size_t)(mBase + ar) * K + k0 + ac;
#pragma unroll
      for (int q = 0; q < 2; ++q) {
        float4 a0 = *(const float4*)(ap + q * 8);
        float4 a1 = *(const float4*)(ap + q * 8 + 4);
        v8h av;
        av[0] = (_Float16)a0.x; av[1] = (_Float16)a0.y; av[2] = (_Float16)a0.z; av[3] = (_Float16)a0.w;
        av[4] = (_Float16)a1.x; av[5] = (_Float16)a1.y; av[6] = (_Float16)a1.z; av[7] = (_Float16)a1.w;
        *(v8h*)(&Asm[ar][ac + q * 8]) = av;
      }
    }
    // ---- stage W tile (32x128) as f16
    {
      const float* wp = W + (size_t)(k0 + wr) * N + nBase + wc;
#pragma unroll
      for (int q = 0; q < 2; ++q) {
        float4 w0 = *(const float4*)(wp + q * 8);
        float4 w1 = *(const float4*)(wp + q * 8 + 4);
        v8h wv;
        wv[0] = (_Float16)w0.x; wv[1] = (_Float16)w0.y; wv[2] = (_Float16)w0.z; wv[3] = (_Float16)w0.w;
        wv[4] = (_Float16)w1.x; wv[5] = (_Float16)w1.y; wv[6] = (_Float16)w1.z; wv[7] = (_Float16)w1.w;
        *(v8h*)(&Bsm[wr][wc + q * 8]) = wv;
      }
      if (k0 + 32 < K)  // hint next tile into cache (global_prefetch_b8)
        __builtin_prefetch((const void*)(wp + (size_t)32 * N), 0, 0);
    }
    __syncthreads();

    v16h a0 = load_a_frag(&Asm[wm * 32][0],      40);
    v16h a1 = load_a_frag(&Asm[wm * 32 + 16][0], 40);
    v16h bf[4];
#pragma unroll
    for (int ni = 0; ni < 4; ++ni)
      bf[ni] = load_b_frag(&Bsm[0][wn * 64 + ni * 16], 136);
#pragma unroll
    for (int ni = 0; ni < 4; ++ni) {
      acc[0][ni] = wmma_f16(a0, bf[ni], acc[0][ni]);
      acc[1][ni] = wmma_f16(a1, bf[ni], acc[1][ni]);
    }
  }

  // ---- epilogue: C layout per ISA (VGPR r -> M = r + 8*(lane>=16), N = lane&15)
  const int col = lane & 15, hi = lane >> 4;
#pragma unroll
  for (int mi = 0; mi < 2; ++mi) {
#pragma unroll
    for (int ni = 0; ni < 4; ++ni) {
#pragma unroll
      for (int r = 0; r < 8; ++r) {
        const int m = mBase + wm * 32 + mi * 16 + r + 8 * hi;
        const int n = nBase + wn * 64 + ni * 16 + col;
        const float v = acc[mi][ni][r];
        if (MODE == 0) {
          _Float16* C = (_Float16*)Cv;
          const int b = m >> 11, s = m & (S_ - 1);
          const int h = n >> 7, d = n & 127;
          C[(((size_t)(b * H_ + h) * S_) + s) * 128 + d] = (_Float16)v;
        } else if (MODE == 1) {
          _Float16* C = (_Float16*)Cv;
          const int b = m >> 11, s = m & (S_ - 1);
          const int h = n >> 6, d = n & 63;
          C[(((size_t)(b * H_ + h) * DH_) + d) * S_ + s] = (_Float16)v;
        } else {
          float* C = (float*)Cv;
          C[(size_t)m * N + n] = v;
        }
      }
    }
  }
}

// --------------------------------------------------------------------------
// Differential attention core. One block = (b,h) x 16 query rows.
// LDS: full s1/s2 score rows (2 x 16 x 2048 f32), exploiting 320KB/WGP.
__launch_bounds__(256, 1)
__global__ void attn_kernel(const _Float16* __restrict__ Qh,
                            const _Float16* __restrict__ Kh,
                            const _Float16* __restrict__ Vt,
                            const float* __restrict__ mask,
                            const float* __restrict__ lam_ws,
                            const float* __restrict__ rms_scale,
                            _Float16* __restrict__ AO) {
  extern __shared__ __align__(16) char smem[];
  float*    s1    = (float*)smem;                 // 16 x 2048
  float*    s2    = s1 + 16 * S_;                 // 16 x 2048
  _Float16* qtile = (_Float16*)(s2 + 16 * S_);    // 16 x 128
  float*    red   = (float*)(qtile + 16 * 128);   // 8 x 16 x 64 partials
  float*    red1  = red + 8 * 16 * 64;            // 16 x 16
  float*    red2  = red1 + 256;                   // 16 x 16
  float*    inv1  = red2 + 256;                   // 16
  float*    inv2  = inv1 + 16;                    // 16

  const int tid  = threadIdx.x;
  const int lane = tid & 31;
  const int wave = tid >> 5;
  const int qt = blockIdx.x;
  const int bh = blockIdx.y;
  const int b  = bh >> 4, h = bh & 15;
  const float lam = lam_ws[h];
  const int col = lane & 15, hi = lane >> 4;

  // stage q tile (16 x 128 halves, contiguous in global)
  {
    const _Float16* src = Qh + ((size_t)bh * S_ + qt * 16) * 128;
    *(v8h*)(qtile + tid * 8) = *(const v8h*)(src + tid * 8);
  }
  __syncthreads();

  const v16h aq1c0 = load_a_frag(qtile + 0,  128);
  const v16h aq1c1 = load_a_frag(qtile + 32, 128);
  const v16h aq2c0 = load_a_frag(qtile + 64, 128);
  const v16h aq2c1 = load_a_frag(qtile + 96, 128);

  // ---- scores: each wave owns key tiles kt = wave, wave+8, ...
  for (int kt = wave; kt < S_ / 16; kt += 8) {
    const _Float16* kb = Kh + ((size_t)bh * S_ + kt * 16 + col) * 128 + 16 * hi;
    v16h bk1c0 = load_b_frag_contig(kb);
    v16h bk1c1 = load_b_frag_contig(kb + 32);
    v16h bk2c0 = load_b_frag_contig(kb + 64);
    v16h bk2c1 = load_b_frag_contig(kb + 96);
    v8f c1 = {}; v8f c2 = {};
    c1 = wmma_f16(aq1c0, bk1c0, c1);
    c1 = wmma_f16(aq1c1, bk1c1, c1);
    c2 = wmma_f16(aq2c0, bk2c0, c2);
    c2 = wmma_f16(aq2c1, bk2c1, c2);
    const float* mb = mask + ((size_t)bh * S_ + qt * 16) * S_ + kt * 16;
#pragma unroll
    for (int r = 0; r < 8; ++r) {
      const int mr = r + 8 * hi;
      const float mv = mb[(size_t)mr * S_ + col];
      s1[mr * S_ + kt * 16 + col] = c1[r] * SCALE_ + mv;
      s2[mr * S_ + kt * 16 + col] = c2[r] * SCALE_ + mv;
    }
  }
  __syncthreads();

  // ---- dual softmax in LDS: 16 threads per row
  {
    const int tsub = tid & 15, row = tid >> 4;
    float m1 = -1e30f, m2 = -1e30f;
    for (int j = tsub; j < S_; j += 16) {
      m1 = fmaxf(m1, s1[row * S_ + j]);
      m2 = fmaxf(m2, s2[row * S_ + j]);
    }
    red1[row * 16 + tsub] = m1;
    red2[row * 16 + tsub] = m2;
    __syncthreads();
    float M1 = -1e30f, M2 = -1e30f;
    for (int i = 0; i < 16; ++i) {
      M1 = fmaxf(M1, red1[row * 16 + i]);
      M2 = fmaxf(M2, red2[row * 16 + i]);
    }
    float sum1 = 0.f, sum2 = 0.f;
    for (int j = tsub; j < S_; j += 16) {
      float p1 = __expf(s1[row * S_ + j] - M1); s1[row * S_ + j] = p1; sum1 += p1;
      float p2 = __expf(s2[row * S_ + j] - M2); s2[row * S_ + j] = p2; sum2 += p2;
    }
    __syncthreads();
    red1[row * 16 + tsub] = sum1;
    red2[row * 16 + tsub] = sum2;
    __syncthreads();
    float T1 = 0.f, T2 = 0.f;
    for (int i = 0; i < 16; ++i) { T1 += red1[row * 16 + i]; T2 += red2[row * 16 + i]; }
    if (tsub == 0) { inv1[row] = 1.0f / T1; inv2[row] = 1.0f / T2; }
    __syncthreads();
  }

  // ---- out = (softmax1 - lam*softmax2) @ V ; wave owns K-chunk of 256 keys
  {
    v8f o0 = {}, o1 = {}, o2 = {}, o3 = {};
    const float fi1 = inv1[col], fi2 = inv2[col];
    const float* p1r = s1 + col * S_;
    const float* p2r = s2 + col * S_;
    const int jb0 = wave * 256;
    for (int it = 0; it < 8; ++it) {
      const int jb = jb0 + it * 32;
      v16h af;
#pragma unroll
      for (int e = 0; e < 16; ++e) {
        const int kk = jb + e + 16 * hi;
        af[e] = (_Float16)(p1r[kk] * fi1 - lam * (p2r[kk] * fi2));
      }
      const _Float16* vb = Vt + ((size_t)bh * DH_ + col) * S_ + jb + 16 * hi;
      o0 = wmma_f16(af, load_b_frag_contig(vb),            o0);
      o1 = wmma_f16(af, load_b_frag_contig(vb + 16 * S_),  o1);
      o2 = wmma_f16(af, load_b_frag_contig(vb + 32 * S_),  o2);
      o3 = wmma_f16(af, load_b_frag_contig(vb + 48 * S_),  o3);
    }
#pragma unroll
    for (int r = 0; r < 8; ++r) {
      const int mr = r + 8 * hi;
      float* dst = red + ((size_t)wave * 16 + mr) * 64 + col;
      dst[0]  = o0[r];
      dst[16] = o1[r];
      dst[32] = o2[r];
      dst[48] = o3[r];
    }
  }
  __syncthreads();

  // ---- cross-wave reduce (16x64), RMS norm, scale, store f16
  for (int idx = tid; idx < 16 * 64; idx += 256) {
    float s = 0.f;
    for (int w = 0; w < 8; ++w) s += red[w * 1024 + idx];
    red[idx] = s;
  }
  __syncthreads();
  {
    const int tsub = tid & 15, row = tid >> 4;
    float ssq = 0.f;
    for (int d = tsub; d < DH_; d += 16) {
      float v = red[row * 64 + d];
      ssq += v * v;
    }
    red1[row * 16 + tsub] = ssq;
    __syncthreads();
    float tot = 0.f;
    for (int i = 0; i < 16; ++i) tot += red1[row * 16 + i];
    const float rstd = rsqrtf(tot * (1.0f / DH_) + 1e-5f);
    for (int d = tsub; d < DH_; d += 16) {
      float v = red[row * 64 + d] * rstd * rms_scale[d] * (1.0f - LAMBDA_INIT_);
      AO[((size_t)b * S_ + qt * 16 + row) * (H_ * DH_) + h * DH_ + d] = (_Float16)v;
    }
  }
}

// --------------------------------------------------------------------------
extern "C" void kernel_launch(void* const* d_in, const int* in_sizes, int n_in,
                              void* d_out, int out_size, void* d_ws, size_t ws_size,
                              hipStream_t stream) {
  (void)in_sizes; (void)n_in; (void)out_size; (void)ws_size;
  const float* query = (const float*)d_in[0];
  const float* key   = (const float*)d_in[1];
  const float* value = (const float*)d_in[2];
  const float* mask  = (const float*)d_in[3];
  const float* W_q   = (const float*)d_in[4];
  const float* W_k   = (const float*)d_in[5];
  const float* W_v   = (const float*)d_in[6];
  const float* W_o   = (const float*)d_in[7];
  const float* lq1   = (const float*)d_in[8];
  const float* lk1   = (const float*)d_in[9];
  const float* lq2   = (const float*)d_in[10];
  const float* lk2   = (const float*)d_in[11];
  const float* rms_scale = (const float*)d_in[12];

  // workspace carve-up (~48 MB)
  char* ws = (char*)d_ws;
  float*    lam = (float*)ws;                                        // 16 f32
  _Float16* Qh  = (_Float16*)(ws + 256);                             // [b][h][s][128]
  _Float16* Kh  = Qh + (size_t)B_ * H_ * S_ * 128;                   // [b][h][s][128]
  _Float16* Vt  = Kh + (size_t)B_ * H_ * S_ * 128;                   // [b][h][d][s]
  _Float16* AO  = Vt + (size_t)B_ * H_ * DH_ * S_;                   // [b][s][h*64+d]

  const dim3 blk(256);

  lam_kernel<<<1, 32, 0, stream>>>(lq1, lk1, lq2, lk2, lam);

  const int M = B_ * S_;  // 4096
  gemm_kernel<0><<<dim3(2 * D_ / 128, M / 128), blk, 0, stream>>>(query, W_q, Qh, M, 2 * D_, D_);
  gemm_kernel<0><<<dim3(2 * D_ / 128, M / 128), blk, 0, stream>>>(key,   W_k, Kh, M, 2 * D_, D_);
  gemm_kernel<1><<<dim3(D_ / 128,     M / 128), blk, 0, stream>>>(value, W_v, Vt, M, D_, D_);

  const size_t smem = (size_t)2 * 16 * S_ * 4    // s1,s2
                    + (size_t)16 * 128 * 2       // qtile
                    + (size_t)8 * 16 * 64 * 4    // partials
                    + 256 * 4 + 256 * 4          // red1, red2
                    + 16 * 4 + 16 * 4;           // inv1, inv2  (~301 KB, < 320 KB/WGP)
  hipFuncSetAttribute((const void*)attn_kernel,
                      hipFuncAttributeMaxDynamicSharedMemorySize, (int)smem);
  attn_kernel<<<dim3(S_ / 16, B_ * H_), blk, smem, stream>>>(Qh, Kh, Vt, mask, lam,
                                                             rms_scale, AO);

  gemm_kernel<2><<<dim3(D_ / 128, M / 128), blk, 0, stream>>>(AO, W_o, d_out, M, D_, D_);
}
